// MultiHeadAttention_7000796692670
// MI455X (gfx1250) — compile-verified, matrix operand order checked by reading, not run on hardware
//
#include <hip/hip_runtime.h>
#include <hip/hip_bf16.h>

// ---------- vector types for WMMA ----------
typedef __attribute__((ext_vector_type(16))) __bf16 v16bf;
typedef __attribute__((ext_vector_type(8)))  __bf16 bf16x8;
typedef __attribute__((ext_vector_type(8)))  float  v8f;

// ---------- optional CDNA5 async global->LDS path ----------
#if defined(__has_builtin)
#  if __has_builtin(__builtin_amdgcn_global_load_async_to_lds_b128) && \
      __has_builtin(__builtin_amdgcn_s_wait_asynccnt)
#    define HAVE_ASYNC_LDS 1
#  endif
#endif
#ifndef HAVE_ASYNC_LDS
#  define HAVE_ASYNC_LDS 0
#endif

#if HAVE_ASYNC_LDS
// The b128 builtin takes pointers to 128-bit payloads:
//   (v4i addrspace(1)* src, v4i addrspace(3)* dst, imm offset, imm cpol)
typedef int v4i_gcc __attribute__((vector_size(16)));
typedef __attribute__((address_space(1))) v4i_gcc* as1_v4i_p;
typedef __attribute__((address_space(3))) v4i_gcc* as3_v4i_p;
// Generic->AS pointers via integer casts: global generic pointers are
// value-identical to AS1; generic LDS pointers carry the LDS offset in the
// low 32 bits (ISA aperture rules), which is what AS3 expects.
__device__ __forceinline__ void async_copy_b128(const void* g, void* l) {
  __builtin_amdgcn_global_load_async_to_lds_b128(
      (as1_v4i_p)(unsigned long long)(size_t)g,
      (as3_v4i_p)(unsigned)(size_t)l, /*offset=*/0, /*cpol=*/0);
}
__device__ __forceinline__ void wait_async0() {
  __builtin_amdgcn_s_wait_asynccnt(0);
}
#else
__device__ __forceinline__ void wait_async0() {}
#endif

union FragU { struct { bf16x8 lo, hi; } s; v16bf v; };

// Load a 16x32 bf16 fragment (A-layout; B-layout mirrors with rows=N) from a
// row-major [16][pitch] tile. Per ISA 7.12.2: lane<16 holds row=lane,
// K 0..7 (VGPR0-3) and 16..23 (VGPR4-7); lane>=16 holds K 8..15 and 24..31.
__device__ __forceinline__ v16bf load_frag(const __bf16* base, int pitch,
                                           int l16, int hl) {
  FragU u;
  const __bf16* p = base + l16 * pitch + hl * 8;
  u.s.lo = *reinterpret_cast<const bf16x8*>(p);
  u.s.hi = *reinterpret_cast<const bf16x8*>(p + 16);
  return u.v;
}

__device__ __forceinline__ v8f wmma_bf16(v16bf a, v16bf b, v8f c) {
  return __builtin_amdgcn_wmma_f32_16x16x32_bf16(
      false, a, false, b, (short)0, c, false, false);
}

__device__ __forceinline__ void store_out(float* p, float v)  { *p = v; }
__device__ __forceinline__ void store_out(__bf16* p, float v) { *p = (__bf16)v; }

// ---------- fp32 -> bf16 cast ----------
__global__ void cast_f32_bf16(const float* __restrict__ src,
                              __bf16* __restrict__ dst, int n) {
  int i = blockIdx.x * blockDim.x + threadIdx.x;
  if (i < n) dst[i] = (__bf16)src[i];
}

// ---------- 128x128x(K) GEMM: C = A[M,K] * B[K,N] (+bias), bf16 in ----------
// Double-buffered LDS staging: A-tile copied with async-to-LDS (overlaps with
// WMMA of the current tile), B-tile register-double-buffered with a post-
// compute transpose scatter into [n][k] layout.
template <typename OutT>
__global__ void __launch_bounds__(256)
gemm_bf16_128x128(const __bf16* __restrict__ A, const __bf16* __restrict__ Bm,
                  OutT* __restrict__ C, const float* __restrict__ bias,
                  int M, int N, int K) {
  __shared__ __align__(16) __bf16 As[2][128 * 32];   // [m][k]
  __shared__ __align__(16) __bf16 Bs[2][128 * 32];   // transposed [n][k]

  const int tid  = threadIdx.x;
  const int wave = tid >> 5;
  const int lane = tid & 31;
  const int hl   = lane >> 4;
  const int l16  = lane & 15;
  const int bm = blockIdx.y * 128;
  const int bn = blockIdx.x * 128;
  const int wm = (wave & 3) * 32;   // 4 wave-rows of 32
  const int wn = (wave >> 2) * 64;  // 2 wave-cols of 64

  v8f acc[2][4] = {};

  // A staging: each thread owns one 16-elem span of one row
  const int arow = tid >> 1;
  const int akc  = (tid & 1) * 16;
  const __bf16* aptr = A + (size_t)(bm + arow) * K + akc;

  // B staging: each thread owns rows bkr and bkr+16, 8 cols at bnc
  const int bkr = tid >> 4;          // 0..15
  const int bnc = (tid & 15) * 8;    // 0..120
  const __bf16* bptr = Bm + (size_t)bkr * N + bn + bnc;

  bf16x8 breg[2];

  // ---- prologue: stage tile 0 into buffer 0 ----
#if HAVE_ASYNC_LDS
  async_copy_b128(aptr,     &As[0][arow * 32 + akc]);
  async_copy_b128(aptr + 8, &As[0][arow * 32 + akc + 8]);
#else
  {
    const bf16x8* srcA = reinterpret_cast<const bf16x8*>(aptr);
    bf16x8* dstA = reinterpret_cast<bf16x8*>(&As[0][arow * 32 + akc]);
    dstA[0] = srcA[0];
    dstA[1] = srcA[1];
  }
#endif
  #pragma unroll
  for (int c = 0; c < 2; ++c)
    breg[c] = *reinterpret_cast<const bf16x8*>(bptr + (size_t)(16 * c) * N);
  #pragma unroll
  for (int c = 0; c < 2; ++c)
    #pragma unroll
    for (int j = 0; j < 8; ++j)
      Bs[0][(bnc + j) * 32 + bkr + 16 * c] = breg[c][j];
  wait_async0();
  __syncthreads();

  int cur = 0;
  for (int k0 = 0; k0 < K; k0 += 32) {
    const int nxt = cur ^ 1;
    const bool more = (k0 + 32) < K;
#if !HAVE_ASYNC_LDS
    bf16x8 areg[2];
#endif
    if (more) {
#if HAVE_ASYNC_LDS
      async_copy_b128(aptr + k0 + 32, &As[nxt][arow * 32 + akc]);
      async_copy_b128(aptr + k0 + 40, &As[nxt][arow * 32 + akc + 8]);
#else
      const bf16x8* srcA = reinterpret_cast<const bf16x8*>(aptr + k0 + 32);
      areg[0] = srcA[0];
      areg[1] = srcA[1];
#endif
      #pragma unroll
      for (int c = 0; c < 2; ++c)
        breg[c] = *reinterpret_cast<const bf16x8*>(
            bptr + (size_t)(k0 + 32 + 16 * c) * N);
    }
    if (k0 + 64 < K)  // global_prefetch_b8 for the tile after next
      __builtin_prefetch(aptr + k0 + 64, 0, 1);

    // ---- compute tile k0 from buffer `cur` ----
    v16bf afrag[2];
    #pragma unroll
    for (int tm = 0; tm < 2; ++tm)
      afrag[tm] = load_frag(&As[cur][(wm + tm * 16) * 32], 32, l16, hl);
    #pragma unroll
    for (int tn = 0; tn < 4; ++tn) {
      v16bf bfrag = load_frag(&Bs[cur][(wn + tn * 16) * 32], 32, l16, hl);
      #pragma unroll
      for (int tm = 0; tm < 2; ++tm)
        acc[tm][tn] = wmma_bf16(afrag[tm], bfrag, acc[tm][tn]);
    }

    // ---- finish staging tile k0+32 into buffer `nxt` ----
    if (more) {
#if !HAVE_ASYNC_LDS
      bf16x8* dstA = reinterpret_cast<bf16x8*>(&As[nxt][arow * 32 + akc]);
      dstA[0] = areg[0];
      dstA[1] = areg[1];
#endif
      #pragma unroll
      for (int c = 0; c < 2; ++c)
        #pragma unroll
        for (int j = 0; j < 8; ++j)
          Bs[nxt][(bnc + j) * 32 + bkr + 16 * c] = breg[c][j];
    }
    wait_async0();
    __syncthreads();
    cur = nxt;
  }

  // C/D layout: lane l, VGPR v -> row = v + 8*hl, col = l16
  #pragma unroll
  for (int tm = 0; tm < 2; ++tm)
    #pragma unroll
    for (int tn = 0; tn < 4; ++tn) {
      int col = bn + wn + tn * 16 + l16;
      float bval = bias ? bias[col] : 0.0f;
      #pragma unroll
      for (int v = 0; v < 8; ++v) {
        int row = bm + wm + tm * 16 + hl * 8 + v;
        store_out(&C[(size_t)row * N + col], acc[tm][tn][v] + bval);
      }
    }
}

// ---------- flash attention: N=1024, d=64, 16 heads, 8 batches ----------
// qkv layout: [b][n][3*1024] bf16, q at +0, k at +1024, v at +2048, head*64.
__global__ void __launch_bounds__(256)
mha_flash(const __bf16* __restrict__ qkv, __bf16* __restrict__ ctx) {
  __shared__ __align__(16) __bf16 Ks[128 * 64];      // [n][d]
  __shared__ __align__(16) __bf16 Vs[64 * 128];      // transposed [d][n]
  __shared__ __align__(16) __bf16 Ps[8][16 * 128];   // per-wave P slice

  const int tid  = threadIdx.x;
  const int wave = tid >> 5;
  const int lane = tid & 31;
  const int hl   = lane >> 4;
  const int l16  = lane & 15;
  const int bId  = blockIdx.z;
  const int head = blockIdx.y;
  const int rbase = blockIdx.x * 128 + wave * 16;   // this wave's 16 Q rows

  const int RP = 3072;  // qkv row pitch
  const __bf16* qbase = qkv + ((size_t)bId * 1024 + rbase) * RP + head * 64;

  // Q fragments straight from global (row-major is fragment-friendly), cached.
  v16bf qf[2];
  #pragma unroll
  for (int ks = 0; ks < 2; ++ks)
    qf[ks] = load_frag(qbase + ks * 32, RP, l16, hl);

  float m[8], lsum[8];
  v8f O[4] = {};
  #pragma unroll
  for (int v = 0; v < 8; ++v) { m[v] = -1e30f; lsum[v] = 0.0f; }
  const float scale = 0.03125f;  // 1024^-0.5 (full dim, per reference)

  const int ldn = tid >> 1;      // row this thread stages
  const int ldh = tid & 1;       // which 32-elem half of d
  const __bf16* kbaseg = qkv + ((size_t)bId * 1024) * RP + 1024 + head * 64 + ldh * 32;
  const __bf16* vbaseg = qkv + ((size_t)bId * 1024) * RP + 2048 + head * 64 + ldh * 32;

  for (int jt = 0; jt < 8; ++jt) {
    const int n0 = jt * 128;
    __syncthreads();
    { // K tile -> Ks[n][d]: async DMA overlaps with V load+transpose below
      const __bf16* src = kbaseg + (size_t)(n0 + ldn) * RP;
      __bf16* dst = &Ks[ldn * 64 + ldh * 32];
#if HAVE_ASYNC_LDS
      #pragma unroll
      for (int c = 0; c < 4; ++c)
        async_copy_b128(src + c * 8, dst + c * 8);
#else
      const bf16x8* s8 = reinterpret_cast<const bf16x8*>(src);
      bf16x8* d8 = reinterpret_cast<bf16x8*>(dst);
      #pragma unroll
      for (int c = 0; c < 4; ++c) d8[c] = s8[c];
#endif
    }
    { // V tile -> Vs[d][n] (transpose via scattered b16 stores)
      const bf16x8* src = reinterpret_cast<const bf16x8*>(
          vbaseg + (size_t)(n0 + ldn) * RP);
      #pragma unroll
      for (int c = 0; c < 4; ++c) {
        bf16x8 vv = src[c];
        #pragma unroll
        for (int j = 0; j < 8; ++j)
          Vs[(ldh * 32 + c * 8 + j) * 128 + ldn] = vv[j];
      }
    }
    wait_async0();
    __syncthreads();

    // S = Q K^T  (B-fragment col = key index -> rows of Ks, k over d)
    v8f S[8] = {};
    #pragma unroll
    for (int tn = 0; tn < 8; ++tn)
      #pragma unroll
      for (int ks = 0; ks < 2; ++ks) {
        v16bf kfrag = load_frag(&Ks[(tn * 16) * 64 + ks * 32], 64, l16, hl);
        S[tn] = wmma_bf16(qf[ks], kfrag, S[tn]);
      }

    // --- online softmax (rows live in 16-lane halves; shfl_xor 1/2/4/8) ---
    float mnew[8];
    #pragma unroll
    for (int v = 0; v < 8; ++v) mnew[v] = m[v];
    #pragma unroll
    for (int tn = 0; tn < 8; ++tn)
      #pragma unroll
      for (int v = 0; v < 8; ++v)
        mnew[v] = fmaxf(mnew[v], S[tn][v] * scale);
    #pragma unroll
    for (int off = 1; off < 16; off <<= 1)
      #pragma unroll
      for (int v = 0; v < 8; ++v)
        mnew[v] = fmaxf(mnew[v], __shfl_xor(mnew[v], off, 32));

    float alpha[8], rsum[8];
    #pragma unroll
    for (int v = 0; v < 8; ++v) {
      alpha[v] = __expf(m[v] - mnew[v]);
      rsum[v] = 0.0f;
      m[v] = mnew[v];
    }

    // P = exp(S - m), stored through wave-private LDS to re-layout C->A
    __bf16* pw = &Ps[wave][0];
    #pragma unroll
    for (int tn = 0; tn < 8; ++tn)
      #pragma unroll
      for (int v = 0; v < 8; ++v) {
        float p = __expf(S[tn][v] * scale - mnew[v]);
        rsum[v] += p;
        pw[(hl * 8 + v) * 128 + tn * 16 + l16] = (__bf16)p;
      }
    #pragma unroll
    for (int off = 1; off < 16; off <<= 1)
      #pragma unroll
      for (int v = 0; v < 8; ++v)
        rsum[v] += __shfl_xor(rsum[v], off, 32);
    #pragma unroll
    for (int v = 0; v < 8; ++v) lsum[v] = lsum[v] * alpha[v] + rsum[v];
    #pragma unroll
    for (int dt = 0; dt < 4; ++dt)
      #pragma unroll
      for (int v = 0; v < 8; ++v)
        O[dt][v] *= alpha[v];

    // O += P V   (B-fragment col = d index -> rows of Vs[d][n], k over n)
    #pragma unroll
    for (int dt = 0; dt < 4; ++dt)
      #pragma unroll
      for (int ks = 0; ks < 4; ++ks) {
        v16bf pf = load_frag(pw + ks * 32, 128, l16, hl);
        v16bf vf = load_frag(&Vs[(dt * 16) * 128 + ks * 32], 128, l16, hl);
        O[dt] = wmma_bf16(pf, vf, O[dt]);
      }
  }

  #pragma unroll
  for (int v = 0; v < 8; ++v) lsum[v] = 1.0f / lsum[v];
  __bf16* obase = ctx + ((size_t)bId * 1024 + rbase) * 1024 + head * 64;
  #pragma unroll
  for (int dt = 0; dt < 4; ++dt)
    #pragma unroll
    for (int v = 0; v < 8; ++v)
      obase[(size_t)(hl * 8 + v) * 1024 + dt * 16 + l16] =
          (__bf16)(O[dt][v] * lsum[v]);
}

// ---------- launch ----------
extern "C" void kernel_launch(void* const* d_in, const int* in_sizes, int n_in,
                              void* d_out, int out_size, void* d_ws, size_t ws_size,
                              hipStream_t stream) {
  (void)in_sizes; (void)n_in; (void)out_size; (void)ws_size;
  const float* x     = (const float*)d_in[0];
  const float* w_qkv = (const float*)d_in[1];
  const float* w_out = (const float*)d_in[2];
  const float* b_out = (const float*)d_in[3];
  float* out = (float*)d_out;

  char* ws = (char*)d_ws;
  const size_t MB = 1024 * 1024;
  __bf16* xb    = (__bf16*)(ws +  0 * MB);  // 16 MB
  __bf16* wqkvb = (__bf16*)(ws + 16 * MB);  //  6 MB
  __bf16* woutb = (__bf16*)(ws + 22 * MB);  //  2 MB
  __bf16* qkvb  = (__bf16*)(ws + 24 * MB);  // 48 MB
  __bf16* ctxb  = (__bf16*)(ws + 72 * MB);  // 16 MB

  const int M = 8192, K = 1024, N3 = 3072, N1 = 1024;

  cast_f32_bf16<<<(M * K  + 255) / 256, 256, 0, stream>>>(x,     xb,    M * K);
  cast_f32_bf16<<<(K * N3 + 255) / 256, 256, 0, stream>>>(w_qkv, wqkvb, K * N3);
  cast_f32_bf16<<<(K * N1 + 255) / 256, 256, 0, stream>>>(w_out, woutb, K * N1);

  gemm_bf16_128x128<__bf16><<<dim3(N3 / 128, M / 128), 256, 0, stream>>>(
      xb, wqkvb, qkvb, nullptr, M, N3, K);

  mha_flash<<<dim3(8, 16, 8), 256, 0, stream>>>(qkvb, ctxb);

  gemm_bf16_128x128<float><<<dim3(N1 / 128, M / 128), 256, 0, stream>>>(
      ctxb, woutb, out, b_out, M, N1, K);
}
